// SlotAttention_7627861918203
// MI455X (gfx1250) — matrix operand order repaired and checked by reading, not compiled, not run on hardware
//
#include <hip/hip_runtime.h>
#include <hip/hip_bf16.h>
#include <math.h>

// Problem constants (match reference)
#define B_    64
#define N_    4096
#define F_    256
#define D_    256
#define KVQ_  256
#define H_    4
#define S_    8
#define DH_   64
#define ITERS_ 3
#define SCALE_  0.125f      // DH^-0.5
#define EPS_    1e-8f
#define LN_EPS_ 1e-5f

typedef __bf16 bhalf;
typedef __attribute__((ext_vector_type(16))) __bf16 bf16x16;
typedef __attribute__((ext_vector_type(8)))  __bf16 bf16x8;
typedef __attribute__((ext_vector_type(8)))  float  f32x8;
typedef __attribute__((ext_vector_type(4)))  unsigned int u32x4;
typedef __attribute__((ext_vector_type(4)))  int i32x4;
typedef __attribute__((ext_vector_type(8)))  int i32x8;

// ---------------- WMMA fragment helpers (CDNA5 wave32; contiguous k order, same in A and B) ---

__device__ __forceinline__ f32x8 f32x8_zero() {
  f32x8 z;
#pragma unroll
  for (int i = 0; i < 8; ++i) z[i] = 0.f;
  return z;
}

// A matrix 16x32 bf16, row-major with leading dim ld.
// lanes 0-15: row=lane, K 0..15; lanes 16-31: row=lane-16, K 16..31
__device__ __forceinline__ bf16x16 frag_a_rowmajor(const bhalf* base, int ld) {
  int lane = threadIdx.x & 31;
  int row = lane & 15;
  int kh  = (lane >> 4) << 4;   // 0 or 16
  const bhalf* p = base + (long)row * ld + kh;
  bf16x16 a;
#pragma unroll
  for (int i = 0; i < 16; ++i) a[i] = p[i];
  return a;
}

// B matrix 32x16 bf16 given as B^T row-major: base[n*ld + k].
// lanes 0-15: col=lane, K 0..15; lanes 16-31: col=lane-16, K 16..31
__device__ __forceinline__ bf16x16 frag_b_ntrans(const bhalf* base, int ld) {
  int lane = threadIdx.x & 31;
  int col = lane & 15;
  int k0  = (lane >> 4) << 4;   // 0 or 16
  const bhalf* p = base + (long)col * ld + k0;
  bf16x16 b;
#pragma unroll
  for (int i = 0; i < 16; ++i) b[i] = p[i];
  return b;
}

__device__ __forceinline__ f32x8 wmma_bf16(bf16x16 a, bf16x16 b, f32x8 c) {
  return __builtin_amdgcn_wmma_f32_16x16x32_bf16(false, a, false, b, (short)0, c,
                                                 false, false);
}

// ---------------- Tensor Data Mover: 2D bf16 tile -> LDS (ISA ch. 7/8) ------
// D# group1 word0: data_size=1 (2B) <<16 | pad_enable<<20 | pad_interval=3
// (pad after every 16 DWORDs = one 32-elem bf16 row) <<22 | pad_amount=3
// (4 DWORDs = 8 bf16) <<25  => LDS row stride of 40 bf16 elements.
#define TDM_FLAGS_BF16_PAD40 0x06D10000u

__device__ __forceinline__ void tdm_load_tile_2d(unsigned lds_off, const void* gptr,
                                                 unsigned tile_d0, unsigned tile_d1,
                                                 unsigned stride_elems,
                                                 unsigned flags_w0) {
  unsigned long long ga = (unsigned long long)gptr;
  const unsigned td0 = stride_elems;   // tensor_dim0 (OOB check): full row
  const unsigned td1 = 0x40000000u;    // tensor_dim1: large, OOB check only
  u32x4 g0;
  g0[0] = (unsigned)__builtin_amdgcn_readfirstlane(1);              // count=1
  g0[1] = (unsigned)__builtin_amdgcn_readfirstlane((int)lds_off);   // lds_addr
  g0[2] = (unsigned)__builtin_amdgcn_readfirstlane((int)(unsigned)ga);
  g0[3] = (unsigned)__builtin_amdgcn_readfirstlane(
      (int)((unsigned)(ga >> 32) | 0x80000000u));                   // type=2
  i32x8 g1;
  g1[0] = __builtin_amdgcn_readfirstlane((int)flags_w0);
  g1[1] = __builtin_amdgcn_readfirstlane((int)(td0 << 16));                // dim0 @48
  g1[2] = __builtin_amdgcn_readfirstlane((int)((td0 >> 16) | (td1 << 16))); // dim1 @80
  g1[3] = __builtin_amdgcn_readfirstlane((int)((td1 >> 16) | (tile_d0 << 16)));
  g1[4] = __builtin_amdgcn_readfirstlane((int)(tile_d1 & 0xFFFFu)); // tile_dim1,2
  g1[5] = __builtin_amdgcn_readfirstlane((int)stride_elems);        // dim0_stride
  g1[6] = 0;
  g1[7] = 0;
  i32x4 z4;
  z4[0] = z4[1] = z4[2] = z4[3] = 0;   // 2D tensor: groups 2/3 unused
#if defined(__clang_major__) && (__clang_major__ >= 23)
  i32x8 z8;
#pragma unroll
  for (int i = 0; i < 8; ++i) z8[i] = 0;
  __builtin_amdgcn_tensor_load_to_lds(g0, g1, z4, z4, z8, 0);
#else
  __builtin_amdgcn_tensor_load_to_lds(g0, g1, z4, z4, 0);
#endif
}

// ---------------- small utility kernels ------------------------------------

__global__ void k_zero_bf16(bhalf* p, long n) {
  long i = (long)blockIdx.x * 256 + threadIdx.x;
  if (i < n) p[i] = (bhalf)0.f;
}

__global__ void k_zero_f32(float* p, int n) {
  int i = blockIdx.x * 256 + threadIdx.x;
  if (i < n) p[i] = 0.f;
}

// convert weights to bf16; fold SCALE into wq
__global__ void k_wcvt(const float* wq, const float* wk, const float* wv,
                       bhalf* wqb, bhalf* wkb, bhalf* wvb) {
  int i = blockIdx.x * 256 + threadIdx.x;
  if (i < KVQ_ * D_) {
    wqb[i] = (bhalf)(wq[i] * SCALE_);
    wkb[i] = (bhalf)wk[i];
    wvb[i] = (bhalf)wv[i];
  }
}

__device__ __forceinline__ float block_sum256(float v, float* sbuf) {
  int t = threadIdx.x;
#pragma unroll
  for (int o = 16; o > 0; o >>= 1) v += __shfl_down(v, o, 32);
  if ((t & 31) == 0) sbuf[t >> 5] = v;
  __syncthreads();
  float total = 0.f;
#pragma unroll
  for (int i = 0; i < 8; ++i) total += sbuf[i];
  __syncthreads();
  return total;
}

// LayerNorm of inputs [B*N, 256] -> bf16
__global__ void k_ln_in(const float* __restrict__ x, const float* __restrict__ w,
                        const float* __restrict__ bia, bhalf* __restrict__ out) {
  __shared__ float sbuf[8];
  long row = blockIdx.x;
  int t = threadIdx.x;
  float v = x[row * 256 + t];
  float m = block_sum256(v, sbuf) * (1.f / 256.f);
  float d = v - m;
  float var = block_sum256(d * d, sbuf) * (1.f / 256.f);
  out[row * 256 + t] = (bhalf)(d * rsqrtf(var + LN_EPS_) * w[t] + bia[t]);
}

// ---------------- K/V projection GEMM (WMMA + TDM double buffering) ---------
// C[M,256] = X[M,256] * W^T, 128x128 block tile, K-step 32.  A/B tiles are
// DMA'd by the Tensor Data Mover into padded LDS (stride 40), double buffered
// so the DMA overlaps the 8 WMMA issues per wave per K-step.
__global__ void k_proj(const bhalf* __restrict__ xb, const bhalf* __restrict__ wkb,
                       const bhalf* __restrict__ wvb, bhalf* __restrict__ kb,
                       bhalf* __restrict__ vb) {
  __shared__ bhalf lA[2][128 * 40];
  __shared__ bhalf lB[2][128 * 40];
  const bhalf* wb = (blockIdx.z == 0) ? wkb : wvb;
  bhalf* ob       = (blockIdx.z == 0) ? kb : vb;
  const long m0 = (long)blockIdx.x * 128;
  const int n0 = blockIdx.y * 128;
  const int t = threadIdx.x;
  const int w = t >> 5, lane = t & 31;
  const int wm = w >> 1, wn = w & 1;   // waves in 4x2 grid over the 128x128 tile

  f32x8 acc[2][4];
#pragma unroll
  for (int i = 0; i < 2; ++i)
#pragma unroll
    for (int j = 0; j < 4; ++j) acc[i][j] = f32x8_zero();

  if (w == 0) {   // prologue: DMA first K-tile pair into buffer 0
    tdm_load_tile_2d((unsigned)(size_t)&lA[0][0], xb + m0 * 256, 32, 128, 256,
                     TDM_FLAGS_BF16_PAD40);
    tdm_load_tile_2d((unsigned)(size_t)&lB[0][0], wb + (long)n0 * 256, 32, 128, 256,
                     TDM_FLAGS_BF16_PAD40);
  }

  for (int kk = 0; kk < 256; kk += 32) {
    const int buf = (kk >> 5) & 1;
    __syncthreads();   // prior compute on buf^1 finished -> safe to overwrite
    if (w == 0) {
      if (kk + 32 < 256) {
        tdm_load_tile_2d((unsigned)(size_t)&lA[buf ^ 1][0], xb + m0 * 256 + kk + 32,
                         32, 128, 256, TDM_FLAGS_BF16_PAD40);
        tdm_load_tile_2d((unsigned)(size_t)&lB[buf ^ 1][0],
                         wb + (long)n0 * 256 + kk + 32, 32, 128, 256,
                         TDM_FLAGS_BF16_PAD40);
        __builtin_amdgcn_s_wait_tensorcnt(2);   // current pair resident
      } else {
        __builtin_amdgcn_s_wait_tensorcnt(0);
      }
    }
    __syncthreads();   // buf ready for all waves

    bf16x16 afr[2], bfr[4];
#pragma unroll
    for (int i = 0; i < 2; ++i)
      afr[i] = frag_a_rowmajor(&lA[buf][(wm * 32 + i * 16) * 40], 40);
#pragma unroll
    for (int j = 0; j < 4; ++j)
      bfr[j] = frag_b_ntrans(&lB[buf][(wn * 64 + j * 16) * 40], 40);
#pragma unroll
    for (int i = 0; i < 2; ++i)
#pragma unroll
      for (int j = 0; j < 4; ++j) acc[i][j] = wmma_bf16(afr[i], bfr[j], acc[i][j]);
  }

  const int col = lane & 15, row0 = (lane >> 4) * 8;
#pragma unroll
  for (int i = 0; i < 2; ++i)
#pragma unroll
    for (int j = 0; j < 4; ++j) {
      long mrow = m0 + wm * 32 + i * 16 + row0;
#pragma unroll
      for (int rr = 0; rr < 8; ++rr)
        ob[(mrow + rr) * 256 + n0 + wn * 64 + j * 16 + col] = (bhalf)acc[i][j][rr];
    }
}

// ---------------- LN(slots) + Q projection (tiny, scalar) -------------------
__global__ void k_lnq(const float* __restrict__ slots, const float* __restrict__ w,
                      const float* __restrict__ bia, const bhalf* __restrict__ wqb,
                      bhalf* __restrict__ qb) {
  __shared__ float sbuf[8];
  __shared__ float sn[256];
  int row = blockIdx.x;            // b*8+s
  int b = row >> 3, s = row & 7;
  int t = threadIdx.x;
  float v = slots[(long)row * 256 + t];
  float m = block_sum256(v, sbuf) * (1.f / 256.f);
  float d = v - m;
  float var = block_sum256(d * d, sbuf) * (1.f / 256.f);
  sn[t] = d * rsqrtf(var + LN_EPS_) * w[t] + bia[t];
  __syncthreads();
  float q = 0.f;
  const bhalf* wr = wqb + (long)t * 256;   // wq row t (out channel), contiguous
#pragma unroll 4
  for (int f = 0; f < 256; ++f) q += sn[f] * (float)wr[f];
  int h = t >> 6, dd = t & 63;
  // padded 16-row layout per (b,h); rows 8..15 pre-zeroed
  qb[((long)(b * 4 + h) * 16 + s) * 64 + dd] = (bhalf)q;
}

// ---------------- dots GEMM: per (b,h) C[16,N] = Q[16,64] * K^T -------------
__global__ void k_dots(const bhalf* __restrict__ qb, const bhalf* __restrict__ kb,
                       float* __restrict__ dots) {
  int bh = blockIdx.y;
  int b = bh >> 2, h = bh & 3;
  int t = threadIdx.x, w = t >> 5, lane = t & 31;
  int j0 = blockIdx.x * 128 + w * 16;
  int col = lane & 15, ksel = (lane >> 4) << 4;

  // streaming prefetch of the next j-tile of K (-> global_prefetch_b8)
  __builtin_prefetch(kb + ((long)b * N_ + ((j0 + 128) & (N_ - 1)) + col) * 256 + h * 64,
                     0, 1);

  f32x8 acc = f32x8_zero();
  const bhalf* qbase = qb + (long)bh * 16 * 64;
#pragma unroll
  for (int kk = 0; kk < 64; kk += 32) {
    bf16x16 a = frag_a_rowmajor(qbase + kk, 64);
    // B^T[n=j][k=d] = K[b, j, h*64+d]: 16 contiguous bf16 per lane (32B load)
    bf16x16 bfr =
        *(const bf16x16*)(kb + ((long)b * N_ + j0 + col) * 256 + h * 64 + kk + ksel);
    acc = wmma_bf16(a, bfr, acc);
  }
  if ((lane >> 4) == 0) {   // rows 0..7 valid (rows 8..15 are zero padding)
#pragma unroll
    for (int rr = 0; rr < 8; ++rr)
      dots[((long)b * 32 + h * 8 + rr) * N_ + j0 + col] = acc[rr];
  }
}

// ---------------- inverted softmax over (s,h)=32, + row sums ----------------
__global__ void k_softmax(const float* __restrict__ dots, bhalf* __restrict__ attnb,
                          float* __restrict__ rowsum, float* __restrict__ attn_out,
                          int write_out) {
  __shared__ float ls[32];
  int b = blockIdx.x >> 4;                      // 16 blocks of 256 j per batch
  int j = ((blockIdx.x & 15) << 8) + threadIdx.x;
  int t = threadIdx.x, lane = t & 31;

  float av[32];
  float mx = -1e30f;
#pragma unroll
  for (int r = 0; r < 32; ++r) {
    av[r] = dots[((long)b * 32 + r) * N_ + j];
    mx = fmaxf(mx, av[r]);
  }
  float sum = 0.f;
#pragma unroll
  for (int r = 0; r < 32; ++r) { av[r] = __expf(av[r] - mx); sum += av[r]; }
  float inv = 1.f / sum;
#pragma unroll
  for (int r = 0; r < 32; ++r) av[r] *= inv;

  if (t < 32) ls[t] = 0.f;
  __syncthreads();
#pragma unroll
  for (int r = 0; r < 32; ++r) {
    float s = av[r];
#pragma unroll
    for (int o = 16; o > 0; o >>= 1) s += __shfl_down(s, o, 32);
    if (lane == 0) atomicAdd(&ls[r], s);
  }
#pragma unroll
  for (int r = 0; r < 32; ++r) {
    int h = r >> 3, ss = r & 7;
    attnb[((long)(b * 4 + h) * 16 + ss) * N_ + j] = (bhalf)(av[r] + EPS_);
  }
  if (write_out) {   // mean over heads of pre-EPS attention (final iteration)
#pragma unroll
    for (int ss = 0; ss < 8; ++ss) {
      float mv = 0.25f * (av[ss] + av[8 + ss] + av[16 + ss] + av[24 + ss]);
      attn_out[((long)b * 8 + ss) * N_ + j] = mv;
    }
  }
  __syncthreads();
  if (t < 32) atomicAdd(&rowsum[b * 32 + t], ls[t]);
}

// ---------------- updates GEMM: per (b,h) C[16,64] = A[16,N] * V[N,64] ------
// K split over 8 waves (512 j each), V transposed through LDS, cross-wave LDS
// reduction, row normalization folded into the final store.
__global__ void k_upd(const bhalf* __restrict__ attnb, const bhalf* __restrict__ vb,
                      const float* __restrict__ rowsum, float* __restrict__ updates) {
  __shared__ __align__(32) unsigned char smem[64 * 264 * 2];  // 33792 B
  bhalf* lV  = (bhalf*)smem;   // [64 d][264 (256 j + pad)]
  float* red = (float*)smem;   // reused: 8 waves * 16*64 f32 = 32768 B

  int bh = blockIdx.x;
  int b = bh >> 2, h = bh & 3;
  int t = threadIdx.x, w = t >> 5, lane = t & 31;

  f32x8 acc[4];
#pragma unroll
  for (int dt = 0; dt < 4; ++dt) acc[dt] = f32x8_zero();

  const bhalf* abase = attnb + (long)bh * 16 * N_;
  for (int jc = 0; jc < N_; jc += 256) {
    // prefetch next V chunk while this one is consumed
    __builtin_prefetch(vb + ((long)b * N_ + ((jc + 256) & (N_ - 1)) + t) * 256 + h * 64,
                       0, 1);
    // stage V chunk [256 j][64 d] transposed into LDS as [d][j]
    const bf16x16* vp = (const bf16x16*)(vb + ((long)b * N_ + jc + t) * 256 + h * 64);
#pragma unroll
    for (int blk = 0; blk < 4; ++blk) {
      bf16x16 x = vp[blk];
#pragma unroll
      for (int i = 0; i < 16; ++i) lV[(blk * 16 + i) * 264 + t] = x[i];
    }
    __syncthreads();
    int jsub = w * 32;   // each wave consumes its own 32-j slice of the chunk
    bf16x16 a = frag_a_rowmajor(abase + jc + jsub, N_);
#pragma unroll
    for (int dt = 0; dt < 4; ++dt) {
      bf16x16 bfr = frag_b_ntrans(lV + dt * 16 * 264 + jsub, 264);
      acc[dt] = wmma_bf16(a, bfr, acc[dt]);
    }
    __syncthreads();
  }

  // cross-wave reduction of the eight partial C[16,64] tiles
  {
    int col = lane & 15, row0 = (lane >> 4) * 8;
#pragma unroll
    for (int dt = 0; dt < 4; ++dt)
#pragma unroll
      for (int rr = 0; rr < 8; ++rr)
        red[w * 1024 + (row0 + rr) * 64 + dt * 16 + col] = acc[dt][rr];
  }
  __syncthreads();
  const float nd = (float)N_ * EPS_;
  for (int idx = t; idx < 1024; idx += 256) {
    float s = 0.f;
#pragma unroll
    for (int ww = 0; ww < 8; ++ww) s += red[ww * 1024 + idx];
    int row = idx >> 6, d = idx & 63;
    if (row < 8) {
      float denom = rowsum[b * 32 + h * 8 + row] + nd;
      updates[((long)b * 8 + row) * 256 + h * 64 + d] = s / denom;
    }
  }
}

// ---------------- GRU cell (scalar; 0.4 GFLOP total) ------------------------
__global__ void k_gru(const float* __restrict__ updates, const float* __restrict__ hprev,
                      const float* __restrict__ wih, const float* __restrict__ whh,
                      const float* __restrict__ bih, const float* __restrict__ bhh,
                      float* __restrict__ slots_out, float* __restrict__ final_out,
                      int last) {
  __shared__ float su[256], sh[256];
  long row = blockIdx.x;
  int t = threadIdx.x;
  su[t] = updates[row * 256 + t];
  sh[t] = hprev[row * 256 + t];
  __syncthreads();
  float ir = bih[t], iz = bih[256 + t], in_ = bih[512 + t];
  float hr = bhh[t], hz = bhh[256 + t], hn = bhh[512 + t];
  const float* wr0 = wih + (long)t * 256;
  const float* wz0 = wih + (long)(256 + t) * 256;
  const float* wn0 = wih + (long)(512 + t) * 256;
  const float* vr0 = whh + (long)t * 256;
  const float* vz0 = whh + (long)(256 + t) * 256;
  const float* vn0 = whh + (long)(512 + t) * 256;
#pragma unroll 4
  for (int f = 0; f < 256; ++f) {
    float u = su[f], hp = sh[f];
    ir += u * wr0[f];  iz += u * wz0[f];  in_ += u * wn0[f];
    hr += hp * vr0[f]; hz += hp * vz0[f]; hn += hp * vn0[f];
  }
  float r = 1.f / (1.f + __expf(-(ir + hr)));
  float z = 1.f / (1.f + __expf(-(iz + hz)));
  float n = tanhf(in_ + r * hn);
  float hnew = (1.f - z) * n + z * sh[t];
  slots_out[row * 256 + t] = hnew;
  if (last) final_out[row * 256 + t] = hnew;
}

// ---------------- launch ----------------------------------------------------

extern "C" void kernel_launch(void* const* d_in, const int* in_sizes, int n_in,
                              void* d_out, int out_size, void* d_ws, size_t ws_size,
                              hipStream_t stream) {
  (void)in_sizes; (void)n_in; (void)out_size; (void)ws_size;

  const float* inputs = (const float*)d_in[0];
  const float* cond   = (const float*)d_in[1];
  const float* ln_in_w = (const float*)d_in[2];
  const float* ln_in_b = (const float*)d_in[3];
  const float* ln_s_w  = (const float*)d_in[4];
  const float* ln_s_b  = (const float*)d_in[5];
  const float* wq = (const float*)d_in[6];
  const float* wk = (const float*)d_in[7];
  const float* wv = (const float*)d_in[8];
  const float* gru_wih = (const float*)d_in[9];
  const float* gru_whh = (const float*)d_in[10];
  const float* gru_bih = (const float*)d_in[11];
  const float* gru_bhh = (const float*)d_in[12];

  float* out_slots = (float*)d_out;                       // [B,S,D]
  float* out_attn  = (float*)d_out + (long)B_ * S_ * D_;  // [B,S,N]

  // workspace carve-up (all offsets 256B aligned)
  size_t off = 0;
  auto carve = [&](size_t bytes) {
    void* p = (char*)d_ws + off;
    off += (bytes + 255) & ~(size_t)255;
    return p;
  };
  bhalf* x_bf  = (bhalf*)carve((size_t)B_ * N_ * F_ * 2);     // 134 MB
  bhalf* k_bf  = (bhalf*)carve((size_t)B_ * N_ * KVQ_ * 2);   // 134 MB
  bhalf* v_bf  = (bhalf*)carve((size_t)B_ * N_ * KVQ_ * 2);   // 134 MB
  bhalf* wq_bf = (bhalf*)carve((size_t)KVQ_ * D_ * 2);
  bhalf* wk_bf = (bhalf*)carve((size_t)KVQ_ * F_ * 2);
  bhalf* wv_bf = (bhalf*)carve((size_t)KVQ_ * F_ * 2);
  bhalf* q_bf  = (bhalf*)carve((size_t)B_ * H_ * 16 * DH_ * 2);   // padded M
  float* dots  = (float*)carve((size_t)B_ * 32 * N_ * 4);     // 33.5 MB
  bhalf* at_bf = (bhalf*)carve((size_t)B_ * H_ * 16 * N_ * 2); // padded M, 33.5 MB
  float* rowsum = (float*)carve((size_t)B_ * 32 * 4);
  float* updates = (float*)carve((size_t)B_ * S_ * KVQ_ * 4);
  float* slots   = (float*)carve((size_t)B_ * S_ * D_ * 4);

  // one-time: zero padded rows of q/attn (ws is poisoned)
  {
    long nq = (long)B_ * H_ * 16 * DH_;
    k_zero_bf16<<<dim3((nq + 255) / 256), dim3(256), 0, stream>>>(q_bf, nq);
    long na = (long)B_ * H_ * 16 * N_;
    k_zero_bf16<<<dim3((na + 255) / 256), dim3(256), 0, stream>>>(at_bf, na);
  }
  k_wcvt<<<dim3(256), dim3(256), 0, stream>>>(wq, wk, wv, wq_bf, wk_bf, wv_bf);
  k_ln_in<<<dim3(B_ * N_), dim3(256), 0, stream>>>(inputs, ln_in_w, ln_in_b, x_bf);
  k_proj<<<dim3(B_ * N_ / 128, 2, 2), dim3(256), 0, stream>>>(x_bf, wk_bf, wv_bf,
                                                              k_bf, v_bf);

  const float* slots_src = cond;
  for (int it = 0; it < ITERS_; ++it) {
    int last = (it == ITERS_ - 1);
    k_zero_f32<<<dim3(8), dim3(256), 0, stream>>>(rowsum, B_ * 32);
    k_lnq<<<dim3(B_ * S_), dim3(256), 0, stream>>>(slots_src, ln_s_w, ln_s_b,
                                                   wq_bf, q_bf);
    k_dots<<<dim3(N_ / 128, B_ * H_), dim3(256), 0, stream>>>(q_bf, k_bf, dots);
    k_softmax<<<dim3(B_ * (N_ / 256)), dim3(256), 0, stream>>>(dots, at_bf, rowsum,
                                                               out_attn, last);
    k_upd<<<dim3(B_ * H_), dim3(256), 0, stream>>>(at_bf, v_bf, rowsum, updates);
    k_gru<<<dim3(B_ * S_), dim3(256), 0, stream>>>(updates, slots_src, gru_wih,
                                                   gru_whh, gru_bih, gru_bhh,
                                                   slots, out_slots, last);
    slots_src = slots;
  }
}
